// Seq2Seq1_65077344469395
// MI455X (gfx1250) — matrix operand order repaired and checked by reading, not compiled, not run on hardware
//
#include <hip/hip_runtime.h>
#include <hip/hip_bf16.h>

typedef __attribute__((ext_vector_type(16))) __bf16 v16bf;
typedef __attribute__((ext_vector_type(8)))  __bf16 v8bf;
typedef __attribute__((ext_vector_type(8)))  float  v8f;
typedef __attribute__((ext_vector_type(4)))  int    v4i;

#define AS1 __attribute__((address_space(1)))
#define AS3 __attribute__((address_space(3)))

namespace {
constexpr int B_     = 64;
constexpr int T_IN   = 512;
constexpr int F_     = 256;
constexpr int H_     = 512;
constexpr int T_OUTc = 256;
constexpr int G_     = 4 * H_;            // 2048 gate columns
constexpr int K1     = H_ + F_;           // 768: encoder K = [h | x_t]
constexpr int K2     = H_;                // 512: decoder K
constexpr int NBLK   = 16;                // persistent grid (all resident)
constexpr int NTHR   = 256;               // 8 waves/block
constexpr int NTHREADS = NBLK * NTHR;     // 4096
constexpr int WPB    = NTHR / 32;         // 8 waves per block
constexpr int COLS_PER_BLK = WPB * 16;    // 128 permuted gate columns / block
constexpr int HL_PER_BLK   = COLS_PER_BLK / 4;  // 32 hidden units / block
constexpr int EPT    = (B_ * HL_PER_BLK) / NTHR; // 8 cell elements / thread
constexpr size_t WLDS_BYTES = (size_t)COLS_PER_BLK * K1 * sizeof(__bf16); // 192 KB
constexpr size_t GLDS_OFF   = WLDS_BYTES;
constexpr size_t LDS_BYTES  = WLDS_BYTES + (size_t)B_ * COLS_PER_BLK * sizeof(float); // 224 KB
}

// ---------------- grid-wide barrier (workspace atomics) ----------------
__device__ __forceinline__ void grid_sync(volatile unsigned* bar) {
  __threadfence();
  __syncthreads();
  if (threadIdx.x == 0) {
    unsigned gen = bar[1];
    if (atomicAdd((unsigned*)&bar[0], 1u) == (unsigned)(NBLK - 1)) {
      bar[0] = 0u;
      __threadfence();
      bar[1] = gen + 1u;
    } else {
      while (bar[1] == gen) { __builtin_amdgcn_s_sleep(1); }
    }
  }
  __syncthreads();
  __threadfence();
}

__device__ __forceinline__ float sigmoidf_(float x) {
  return 1.0f / (1.0f + __expf(-x));
}

// stage this block's weight strip (COLS_PER_BLK rows x K bf16) into LDS
__device__ __forceinline__ void stage_weights(__bf16* wlds, const __bf16* wcatp, int K) {
  const v8bf* gsrc = (const v8bf*)(wcatp + (size_t)blockIdx.x * COLS_PER_BLK * K);
  v8bf* ldst = (v8bf*)wlds;
  const int nchunk = COLS_PER_BLK * K / 8;
#if __has_builtin(__builtin_amdgcn_global_load_async_to_lds_b128)
  for (int i = threadIdx.x; i < nchunk; i += NTHR) {
    __builtin_amdgcn_global_load_async_to_lds_b128(
        (AS1 v4i*)(gsrc + i), (AS3 v4i*)(ldst + i), 0, 0);
  }
#if __has_builtin(__builtin_amdgcn_s_wait_asynccnt)
  __builtin_amdgcn_s_wait_asynccnt(0);
#else
  asm volatile("s_wait_asynccnt 0x0" ::: "memory");
#endif
#else
  for (int i = threadIdx.x; i < nchunk; i += NTHR) ldst[i] = gsrc[i];
#endif
  __syncthreads();
}

// ---------------- one LSTM layer on the persistent grid ----------------
// Gate columns are PERMUTED: p = 4*h' + {i,f,g,o}, so each block's 128-column
// strip covers 32 complete hidden units -> elementwise is block-local via LDS.
template <bool HAS_X>
__device__ void lstm_layer(char* smem,
                           const __bf16* __restrict__ wcatp, // [G_ x K] permuted rows
                           const float*  __restrict__ biasp, // [G_] permuted b_ih+b_hh
                           const __bf16* __restrict__ xbf,   // [B_,T_IN,F_] bf16
                           const __bf16* hbuf0, const __bf16* hbuf1, // double buffer
                           float* creg,                      // [EPT] register cell state
                           float* __restrict__ y,            // decoder output
                           volatile unsigned* bar, int T) {
  const int K    = HAS_X ? K1 : K2;
  __bf16* wlds = (__bf16*)smem;
  float*  glds = (float*)(smem + GLDS_OFF);  // [64 rows x 128 cols] gate panel
  const int lane = threadIdx.x & 31;
  const int wv   = threadIdx.x >> 5;
  const int ln   = lane & 15;
  const bool hi  = lane >= 16;

  stage_weights(wlds, wcatp, K);

  // preload per-thread gate biases into registers
  float bi[EPT], bfv[EPT], bgv[EPT], bov[EPT];
#pragma unroll
  for (int r = 0; r < EPT; ++r) {
    int el = threadIdx.x + r * NTHR;
    int hl = el & (HL_PER_BLK - 1);
    const float* bp = biasp + blockIdx.x * COLS_PER_BLK + hl * 4;
    bi[r] = bp[0]; bfv[r] = bp[1]; bgv[r] = bp[2]; bov[r] = bp[3];
  }

  // B fragment source (LDS): row n=wv*16+ln, contiguous K; lo/hi lane K halves
  const __bf16* wrow = wlds + (size_t)(wv * 16 + ln) * K + (hi ? 16 : 0);

  for (int t = 0; t < T; ++t) {
    const __bf16* hread = (t & 1) ? hbuf1 : hbuf0;
    __bf16* hwrite = (__bf16*)((t & 1) ? hbuf0 : hbuf1);

    // ---- GEMM: gates[0:64, strip] = [h | x_t] @ Wp[strip, :]^T ----
    v8f acc[4] = {};
    for (int kt = 0; kt < K; kt += 32) {
      v16bf bfrag = *(const v16bf*)(wrow + kt);        // ds_load_b128 x2
      const int ka = kt + (hi ? 16 : 0);
#pragma unroll
      for (int mt = 0; mt < 4; ++mt) {
        const int m = mt * 16 + ln;
        const __bf16* src;
        if (HAS_X && kt >= H_)
          src = xbf + (size_t)m * (T_IN * F_) + (size_t)t * F_ - H_;
        else
          src = hread + (size_t)m * H_;
        v8bf r0 = *(const v8bf*)(src + ka);
        v8bf r1 = *(const v8bf*)(src + ka + 8);
        v16bf afrag;
#pragma unroll
        for (int i = 0; i < 8; ++i) { afrag[i] = r0[i]; afrag[i + 8] = r1[i]; }
        acc[mt] = __builtin_amdgcn_wmma_f32_16x16x32_bf16(
            false, afrag, false, bfrag, (short)0, acc[mt], false, false);
      }
    }
    // scatter C tiles into the block-local LDS gate panel
#pragma unroll
    for (int mt = 0; mt < 4; ++mt) {
#pragma unroll
      for (int g = 0; g < 8; ++g) {
        int row = mt * 16 + g + (hi ? 8 : 0);
        glds[row * COLS_PER_BLK + wv * 16 + ln] = acc[mt][g];
      }
    }
    __syncthreads();

    // ---- elementwise LSTM cell: block-local hidden units, c in registers ----
#pragma unroll
    for (int r = 0; r < EPT; ++r) {
      int el = threadIdx.x + r * NTHR;
      int b  = el >> 5;                      // batch row
      int hl = el & (HL_PER_BLK - 1);        // local hidden unit
      const float* gp = glds + b * COLS_PER_BLK + hl * 4;
      float ig = sigmoidf_(gp[0] + bi[r]);
      float fg = sigmoidf_(gp[1] + bfv[r]);
      float gg = tanhf    (gp[2] + bgv[r]);
      float og = sigmoidf_(gp[3] + bov[r]);
      float c  = fg * creg[r] + ig * gg;
      creg[r]  = c;
      float hv = og * tanhf(c);
      int hglob = blockIdx.x * HL_PER_BLK + hl;
      hwrite[(size_t)b * H_ + hglob] = (__bf16)hv;
      if (!HAS_X) y[(size_t)b * (T_OUTc * H_) + (size_t)t * H_ + hglob] = hv;
      if (HAS_X && t + 1 < T)   // warm L0/L2 for next step's x rows
        __builtin_prefetch(xbf + (size_t)b * (T_IN * F_) + (size_t)(t + 1) * F_ + hl * 8, 0, 0);
    }
    grid_sync(bar);  // single barrier/step: publish hwrite to all blocks
  }
}

__global__ __launch_bounds__(NTHR) void lstm_persistent(
    const __bf16* __restrict__ xbf,
    const __bf16* __restrict__ wcat1p, const __bf16* __restrict__ wcat2p,
    const float* __restrict__ biasp1,  const float* __restrict__ biasp2,
    __bf16* hbuf0, __bf16* hbuf1, float* y, unsigned* bar) {
  extern __shared__ char smem[];
  float creg[EPT];
#pragma unroll
  for (int r = 0; r < EPT; ++r) creg[r] = 0.0f;
  const int tid = blockIdx.x * NTHR + threadIdx.x;
  for (int e = tid; e < B_ * H_; e += NTHREADS) hbuf0[e] = (__bf16)0.0f;
  grid_sync((volatile unsigned*)bar);
  lstm_layer<true >(smem, wcat1p, biasp1, xbf, hbuf0, hbuf1, creg, nullptr,
                    (volatile unsigned*)bar, T_IN);
  // T_IN even -> parity continues: decoder t=0 reads hbuf0 (encoder's final h)
  lstm_layer<false>(smem, wcat2p, biasp2, nullptr, hbuf0, hbuf1, creg, y,
                    (volatile unsigned*)bar, T_OUTc);
}

// ------------- prep: bf16 convert, gate-permuted weight concat, bias fuse -------------
__global__ void prep_kernel(const float* __restrict__ x,
                            const float* __restrict__ w_ih1, const float* __restrict__ w_hh1,
                            const float* __restrict__ b_ih1, const float* __restrict__ b_hh1,
                            const float* __restrict__ w_hh2,
                            const float* __restrict__ b_ih2, const float* __restrict__ b_hh2,
                            __bf16* xbf, __bf16* wcat1p, __bf16* wcat2p,
                            float* biasp1, float* biasp2, unsigned* bar) {
  size_t tid = (size_t)blockIdx.x * blockDim.x + threadIdx.x;
  size_t nth = (size_t)gridDim.x * blockDim.x;
  for (size_t i = tid; i < (size_t)B_ * T_IN * F_; i += nth)
    xbf[i] = (__bf16)x[i];
  // permuted row p = 4*h' + gate  <-  original row n = gate*H_ + h'
  for (size_t i = tid; i < (size_t)G_ * K1; i += nth) {
    int p = (int)(i / K1), k = (int)(i % K1);
    int n = (p & 3) * H_ + (p >> 2);
    float v = (k < H_) ? w_hh1[(size_t)n * H_ + k] : w_ih1[(size_t)n * F_ + (k - H_)];
    wcat1p[i] = (__bf16)v;
  }
  for (size_t i = tid; i < (size_t)G_ * K2; i += nth) {
    int p = (int)(i / K2), k = (int)(i % K2);
    int n = (p & 3) * H_ + (p >> 2);
    wcat2p[i] = (__bf16)w_hh2[(size_t)n * H_ + k];
  }
  for (size_t i = tid; i < (size_t)G_; i += nth) {
    int n = ((int)i & 3) * H_ + ((int)i >> 2);
    biasp1[i] = b_ih1[n] + b_hh1[n];
    biasp2[i] = b_ih2[n] + b_hh2[n];
  }
  if (tid < 2) bar[tid] = 0u;   // reset grid barrier every launch
}

// ---------------- host launch ----------------
extern "C" void kernel_launch(void* const* d_in, const int* in_sizes, int n_in,
                              void* d_out, int out_size, void* d_ws, size_t ws_size,
                              hipStream_t stream) {
  (void)in_sizes; (void)n_in; (void)out_size; (void)ws_size;
  const float* x     = (const float*)d_in[0];
  const float* w_ih1 = (const float*)d_in[1];
  const float* w_hh1 = (const float*)d_in[2];
  const float* b_ih1 = (const float*)d_in[3];
  const float* b_hh1 = (const float*)d_in[4];
  // d_in[5] = w_ih2: decoder input is all-zero -> mathematically unused
  const float* w_hh2 = (const float*)d_in[6];
  const float* b_ih2 = (const float*)d_in[7];
  const float* b_hh2 = (const float*)d_in[8];
  float* y = (float*)d_out;

  char* ws = (char*)d_ws;
  size_t off = 0;
  auto alloc = [&](size_t bytes) {
    char* p = ws + off;
    off += (bytes + 255) & ~(size_t)255;
    return p;
  };
  __bf16*   xbf    = (__bf16*)  alloc((size_t)B_ * T_IN * F_ * 2);  // 16.8 MB
  __bf16*   wcat1p = (__bf16*)  alloc((size_t)G_ * K1 * 2);         //  3.1 MB
  __bf16*   wcat2p = (__bf16*)  alloc((size_t)G_ * K2 * 2);         //  2.1 MB
  float*    biasp1 = (float*)   alloc((size_t)G_ * 4);
  float*    biasp2 = (float*)   alloc((size_t)G_ * 4);
  __bf16*   hbuf0  = (__bf16*)  alloc((size_t)B_ * H_ * 2);
  __bf16*   hbuf1  = (__bf16*)  alloc((size_t)B_ * H_ * 2);
  unsigned* bar    = (unsigned*)alloc(256);

  prep_kernel<<<2048, 256, 0, stream>>>(x, w_ih1, w_hh1, b_ih1, b_hh1,
                                        w_hh2, b_ih2, b_hh2,
                                        xbf, wcat1p, wcat2p, biasp1, biasp2, bar);
  lstm_persistent<<<NBLK, NTHR, LDS_BYTES, stream>>>(xbf, wcat1p, wcat2p,
                                                     biasp1, biasp2,
                                                     hbuf0, hbuf1, y, bar);
}
